// MultiHeadedAttention_4355096838196
// MI455X (gfx1250) — compile-verified, matrix operand order checked by reading, not run on hardware
//
#include <hip/hip_runtime.h>

// ---------------- problem constants ----------------
#define BATCH 32
#define SEQ   512
#define DMODEL 256
#define NHEAD 8
#define DK    32
#define BS    (BATCH * SEQ)          // 16384 rows
#define SLICES (DMODEL + 1)          // 257 DeCov slices
#define NMAT   (DMODEL * DMODEL)     // 65536 weight elements
#define NACT   (BS * DMODEL)         // 4194304 activation elements

typedef __attribute__((ext_vector_type(16))) __bf16 v16bf;
typedef __attribute__((ext_vector_type(8)))  __bf16 v8bf;
typedef __attribute__((ext_vector_type(8)))  float  v8f;
typedef __attribute__((ext_vector_type(2)))  float  v2f;

#define WMMA_BF16(a, b, c) \
  __builtin_amdgcn_wmma_f32_16x16x32_bf16(false, (a), false, (b), (short)0, (c), false, false)
#define WMMA_F32X4(a, b, c) \
  __builtin_amdgcn_wmma_f32_16x16x4_f32(false, (a), false, (b), (short)0, (c), false, false)

// Row-major [16 rows x 32 K] A/B fragment (ISA 7.12.2, 16-bit 16x32):
// lanes 0-15: regs 0-3 -> K 0..7, regs 4-7 -> K 16..23 ; lanes 16-31: +8.
// K pattern = two contiguous 8-element (16-byte) runs -> two b128 loads.
__device__ __forceinline__ v16bf load_ab_rows_bf16(const __bf16* src, int ld, int lane) {
  int r = lane & 15, hf = lane >> 4;
  const __bf16* p = src + r * ld + 8 * hf;
  v8bf lo = *(const v8bf*)(p);
  v8bf hi = *(const v8bf*)(p + 16);
  return __builtin_shufflevector(lo, hi, 0, 1, 2, 3, 4, 5, 6, 7,
                                 8, 9, 10, 11, 12, 13, 14, 15);
}

// ------------- kernel 0: f32 -> bf16 for weights and the 3 activations -------------
__global__ void cvt_all(const float* Wq, const float* Wk, const float* Wv, const float* Wo,
                        const float* q, const float* k, const float* v,
                        __bf16* wq, __bf16* wk, __bf16* wv, __bf16* wo,
                        __bf16* qx, __bf16* kx, __bf16* vx) {
  int gid = blockIdx.x * blockDim.x + threadIdx.x;
  if (gid < 4 * NMAT) {
    int which = gid >> 16, e = gid & (NMAT - 1);
    const float* s = which == 0 ? Wq : which == 1 ? Wk : which == 2 ? Wv : Wo;
    __bf16* d      = which == 0 ? wq : which == 1 ? wk : which == 2 ? wv : wo;
    d[e] = (__bf16)s[e];
  } else {
    int t = gid - 4 * NMAT;
    int which = t / NACT, e = t - which * NACT;
    const float* s = which == 0 ? q : which == 1 ? k : v;
    __bf16* d      = which == 0 ? qx : which == 1 ? kx : vx;
    d[e] = (__bf16)s[e];
  }
}

// ------------- kernel 1: QKV projection -------------
// out[m,n] = sum_k X[m,k] * W[n,k] + bias[n]   (torch Linear)
// q,k stored [B,H,S,DK];  v stored TRANSPOSED [B,H,DK,S] for fast P@V B-fragments.
__global__ void qkv_proj(const __bf16* qx, const __bf16* kx, const __bf16* vx,
                         const __bf16* wq, const __bf16* wk, const __bf16* wv,
                         const float* bq, const float* bk, const float* bv,
                         __bf16* qws, __bf16* kws, __bf16* vws) {
  int which = blockIdx.z;
  const __bf16* X = which == 0 ? qx : which == 1 ? kx : vx;
  const __bf16* W = which == 0 ? wq : which == 1 ? wk : wv;
  const float* bias = which == 0 ? bq : which == 1 ? bk : bv;
  __bf16* dst = which == 0 ? qws : which == 1 ? kws : vws;

  int m0 = blockIdx.x * 16, n0 = blockIdx.y * 16;
  int lane = threadIdx.x & 31, r = lane & 15, hf = lane >> 4;

  v8f c = {};
#pragma unroll
  for (int kt = 0; kt < DMODEL / 32; ++kt) {
    v16bf a = load_ab_rows_bf16(X + (size_t)m0 * DMODEL + kt * 32, DMODEL, lane);
    v16bf b = load_ab_rows_bf16(W + (size_t)n0 * DMODEL + kt * 32, DMODEL, lane);
    c = WMMA_BF16(a, b, c);
  }
  int n = n0 + r, h = n >> 5, dp = n & 31;
  float bv_ = bias[n];
#pragma unroll
  for (int i = 0; i < 8; ++i) {
    int m = m0 + i + 8 * hf;               // C layout: M = i + 8*(lane/16)
    int bb = m >> 9, s = m & 511;
    float val = c[i] + bv_;
    if (which == 2) {                      // V transposed: [B,H,DK,S]
      dst[(((size_t)bb * NHEAD + h) * DK + dp) * SEQ + s] = (__bf16)val;
    } else {                               // Q/K: [B,H,S,DK]
      dst[(((size_t)bb * NHEAD + h) * SEQ + s) * DK + dp] = (__bf16)val;
    }
  }
}

// ------------- kernel 2: flash attention, one wave = 16 query rows -------------
__global__ void attention(const __bf16* qw, const __bf16* kw, const __bf16* vwT,
                          const int* mask, float* x, __bf16* xbf) {
  __shared__ __bf16 pt[16][32];      // 1 KB: C-layout -> A-layout staging for P

  int bh = blockIdx.x, qt = blockIdx.y;
  int b = bh >> 3, h = bh & 7;
  int lane = threadIdx.x & 31, r = lane & 15, hf = lane >> 4;
  int q0 = qt * 16;

  const __bf16* qb = qw  + (size_t)bh * SEQ * DK;
  const __bf16* kb = kw  + (size_t)bh * SEQ * DK;
  const __bf16* vb = vwT + (size_t)bh * DK * SEQ;   // [DK][SEQ]
  const int* mbase = mask + (size_t)b * SEQ * SEQ + (size_t)q0 * SEQ;

  v16bf aq = load_ab_rows_bf16(qb + q0 * DK, DK, lane);   // 16 x 32, whole dk
  const float scale = 0.17677669529663689f;                // 1/sqrt(32)

  float m_st[8], l_st[8];
#pragma unroll
  for (int i = 0; i < 8; ++i) { m_st[i] = -3.0e38f; l_st[i] = 0.f; }
  v8f c0 = {}, c1 = {};

  for (int kt = 0; kt < SEQ / 32; ++kt) {
    int n0 = kt * 32;
    v16bf bk0 = load_ab_rows_bf16(kb + (size_t)n0 * DK, DK, lane);
    v16bf bk1 = load_ab_rows_bf16(kb + (size_t)(n0 + 16) * DK, DK, lane);
    v8f s0 = {}, s1 = {};
    s0 = WMMA_BF16(aq, bk0, s0);
    s1 = WMMA_BF16(aq, bk1, s1);

    if (kt + 1 < SEQ / 32) {                       // prefetch next k/v tiles
      __builtin_prefetch(kb + (size_t)(n0 + 32) * DK + r * DK, 0, 0);
      __builtin_prefetch(vb + (size_t)r * SEQ + n0 + 32, 0, 0);
    }

    float p0[8], p1[8], alpha[8];
#pragma unroll
    for (int i = 0; i < 8; ++i) {
      int row = i + 8 * hf;
      const int* mr = mbase + row * SEQ + n0 + r;
      float a0 = mr[0]  ? s0[i] * scale : -1e9f;   // scale then mask (as ref)
      float a1 = mr[16] ? s1[i] * scale : -1e9f;
      float t = fmaxf(a0, a1);                     // row max across 16 lanes
      t = fmaxf(t, __shfl_xor(t, 1, 32));
      t = fmaxf(t, __shfl_xor(t, 2, 32));
      t = fmaxf(t, __shfl_xor(t, 4, 32));
      t = fmaxf(t, __shfl_xor(t, 8, 32));
      float mnew = fmaxf(m_st[i], t);
      alpha[i] = __expf(m_st[i] - mnew);
      m_st[i] = mnew;
      p0[i] = __expf(a0 - mnew);
      p1[i] = __expf(a1 - mnew);
      float rs = p0[i] + p1[i];                    // row sum across 16 lanes
      rs += __shfl_xor(rs, 1, 32);
      rs += __shfl_xor(rs, 2, 32);
      rs += __shfl_xor(rs, 4, 32);
      rs += __shfl_xor(rs, 8, 32);
      l_st[i] = alpha[i] * l_st[i] + rs;
    }

    __syncthreads();                               // prev A-frag reads done
#pragma unroll
    for (int i = 0; i < 8; ++i) {
      pt[i + 8 * hf][r]      = (__bf16)p0[i];
      pt[i + 8 * hf][16 + r] = (__bf16)p1[i];
    }
    __syncthreads();                               // stores visible for reads

    v16bf ap  = load_ab_rows_bf16(&pt[0][0], 32, lane);
    // B[k][n] = V[k][n] = vT[n][k]: row-major loader on vT, rows = dk index
    v16bf bv0 = load_ab_rows_bf16(vb + 0 * SEQ + n0,  SEQ, lane);
    v16bf bv1 = load_ab_rows_bf16(vb + 16 * SEQ + n0, SEQ, lane);
#pragma unroll
    for (int i = 0; i < 8; ++i) { c0[i] *= alpha[i]; c1[i] *= alpha[i]; }
    c0 = WMMA_BF16(ap, bv0, c0);
    c1 = WMMA_BF16(ap, bv1, c1);
  }

#pragma unroll
  for (int i = 0; i < 8; ++i) {
    int row = i + 8 * hf;
    float li = 1.0f / l_st[i];
    size_t m = (size_t)b * SEQ + q0 + row;
    float o0 = c0[i] * li, o1 = c1[i] * li;
    x[m * DMODEL + h * DK + r]        = o0;
    x[m * DMODEL + h * DK + 16 + r]   = o1;
    xbf[m * DMODEL + h * DK + r]      = (__bf16)o0;
    xbf[m * DMODEL + h * DK + 16 + r] = (__bf16)o1;
  }
}

// ------------- kernel 3: output projection out = x @ Wo^T + bo -------------
__global__ void out_proj(const __bf16* xbf, const __bf16* wo, const float* bo, float* out) {
  int m0 = blockIdx.x * 16, n0 = blockIdx.y * 16;
  int lane = threadIdx.x & 31, r = lane & 15, hf = lane >> 4;
  v8f c = {};
#pragma unroll
  for (int kt = 0; kt < DMODEL / 32; ++kt) {
    v16bf a = load_ab_rows_bf16(xbf + (size_t)m0 * DMODEL + kt * 32, DMODEL, lane);
    v16bf b = load_ab_rows_bf16(wo + (size_t)n0 * DMODEL + kt * 32, DMODEL, lane);
    c = WMMA_BF16(a, b, c);
  }
  int n = n0 + r;
  float bias = bo[n];
#pragma unroll
  for (int i = 0; i < 8; ++i) {
    int m = m0 + i + 8 * hf;
    out[(size_t)m * DMODEL + n] = c[i] + bias;
  }
}

// ------------- kernel 4a: column means over batch, first 257 seq slices -------------
__global__ void col_means(const float* x, float* mu) {
  int idx = blockIdx.x * blockDim.x + threadIdx.x;
  if (idx >= SLICES * DMODEL) return;
  int s = idx / DMODEL, d = idx - s * DMODEL;
  float acc = 0.f;
#pragma unroll
  for (int b = 0; b < BATCH; ++b)
    acc += x[(size_t)b * SEQ * DMODEL + (size_t)s * DMODEL + d];
  mu[idx] = acc * (1.0f / BATCH);
}

// ------------- kernel 4b: DeCov tiles via f32 WMMA 16x16x4 -> per-block partials -------------
__global__ void decov(const float* x, const float* mu, float* part) {
  int s = blockIdx.x;
  int d0 = blockIdx.y * 16, e0 = blockIdx.z * 16;
  int lane = threadIdx.x & 31, r = lane & 15, hf = lane >> 4;

  const float* mrow = mu + (size_t)s * DMODEL;
  v8f c = {};
#pragma unroll
  for (int step = 0; step < BATCH / 4; ++step) {   // K = batch dim, 4 per WMMA
    v2f a, b;
#pragma unroll
    for (int p = 0; p < 2; ++p) {
      int bb = step * 4 + p + 2 * hf;              // 32-bit A 16x4 layout
      a[p] = x[(size_t)bb * SEQ * DMODEL + (size_t)s * DMODEL + d0 + r] - mrow[d0 + r];
      b[p] = x[(size_t)bb * SEQ * DMODEL + (size_t)s * DMODEL + e0 + r] - mrow[e0 + r];
    }
    c = WMMA_F32X4(a, b, c);
  }
  float acc = 0.f;
#pragma unroll
  for (int i = 0; i < 8; ++i) {
    int d = d0 + i + 8 * hf, e = e0 + r;
    float v = c[i] * (1.0f / (BATCH - 1));
    if (d != e) acc += v * v;                       // fro2 - diag2
  }
#pragma unroll
  for (int m = 16; m >= 1; m >>= 1) acc += __shfl_xor(acc, m, 32);
  if (lane == 0)
    part[(size_t)s * ((DMODEL / 16) * (DMODEL / 16)) + blockIdx.y * (DMODEL / 16) + blockIdx.z]
        = 0.5f * acc;
}

// ------------- kernel 4c: deterministic fixed-order reduction of partials -------------
__global__ void reduce_loss(const float* part, float* loss) {
  __shared__ float red[256];
  int tid = threadIdx.x;
  const int N = SLICES * (DMODEL / 16) * (DMODEL / 16);   // 65792
  float acc = 0.f;
  for (int i = tid; i < N; i += 256) acc += part[i];      // fixed order
  red[tid] = acc;
  __syncthreads();
  for (int s = 128; s > 0; s >>= 1) {
    if (tid < s) red[tid] += red[tid + s];
    __syncthreads();
  }
  if (tid == 0) loss[0] = red[0];
}

// ---------------- launcher ----------------
extern "C" void kernel_launch(void* const* d_in, const int* in_sizes, int n_in,
                              void* d_out, int out_size, void* d_ws, size_t ws_size,
                              hipStream_t stream) {
  const float* query  = (const float*)d_in[0];
  const float* key_in = (const float*)d_in[1];
  const float* value  = (const float*)d_in[2];
  const int*   mask   = (const int*)d_in[3];
  const float* Wq = (const float*)d_in[4];  const float* bq = (const float*)d_in[5];
  const float* Wk = (const float*)d_in[6];  const float* bk = (const float*)d_in[7];
  const float* Wv = (const float*)d_in[8];  const float* bv = (const float*)d_in[9];
  const float* Wo = (const float*)d_in[10]; const float* bo = (const float*)d_in[11];

  float* out  = (float*)d_out;
  float* loss = out + (size_t)BS * DMODEL;     // scalar appended after [B,S,D]

  // workspace partition
  __bf16* wq_bf = (__bf16*)d_ws;
  __bf16* wk_bf = wq_bf + NMAT;
  __bf16* wv_bf = wk_bf + NMAT;
  __bf16* wo_bf = wv_bf + NMAT;
  __bf16* qx    = wo_bf + NMAT;                // bf16 copies of activations
  __bf16* kx    = qx + (size_t)NACT;
  __bf16* vx    = kx + (size_t)NACT;
  __bf16* qws   = vx + (size_t)NACT;           // [B,H,S,DK] bf16
  __bf16* kws   = qws + (size_t)NACT;          // [B,H,S,DK] bf16
  __bf16* vws   = kws + (size_t)NACT;          // [B,H,DK,S] bf16 (transposed)
  __bf16* xbf   = vws + (size_t)NACT;          // [B,S,D] bf16
  float*  xws   = (float*)(xbf + (size_t)NACT);// [B,S,D] f32
  float*  mu    = xws + (size_t)NACT;          // [257,256]
  float*  part  = mu + (size_t)SLICES * DMODEL;// [257*16*16]

  {
    int total = 4 * NMAT + 3 * NACT;
    cvt_all<<<dim3((total + 255) / 256), 256, 0, stream>>>(
        Wq, Wk, Wv, Wo, query, key_in, value,
        wq_bf, wk_bf, wv_bf, wo_bf, qx, kx, vx);
  }

  qkv_proj<<<dim3(BS / 16, DMODEL / 16, 3), 32, 0, stream>>>(
      qx, kx, vx, wq_bf, wk_bf, wv_bf, bq, bk, bv, qws, kws, vws);

  attention<<<dim3(BATCH * NHEAD, SEQ / 16), 32, 0, stream>>>(
      qws, kws, vws, mask, xws, xbf);

  out_proj<<<dim3(BS / 16, DMODEL / 16), 32, 0, stream>>>(xbf, wo_bf, bo, out);

  col_means<<<dim3((SLICES * DMODEL + 255) / 256), 256, 0, stream>>>(xws, mu);

  decov<<<dim3(SLICES, DMODEL / 16, DMODEL / 16), 32, 0, stream>>>(xws, mu, part);

  reduce_loss<<<1, 256, 0, stream>>>(part, loss);
}
